// BitMLP_33500744909529
// MI455X (gfx1250) — 8-bit matrix operand order repaired and checked by reading, compiled, not run on hardware
//
#include <hip/hip_runtime.h>


typedef int v8i __attribute__((ext_vector_type(8)));
typedef int v4i __attribute__((ext_vector_type(4)));

#define BT 128
#define KT 64
#define LDS_STRIDE 80

#if defined(__has_builtin)
#if __has_builtin(__builtin_amdgcn_global_load_async_to_lds_b128) && \
    __has_builtin(__builtin_amdgcn_s_wait_asynccnt)
#define USE_ASYNC_LDS 1
#endif
#endif
#ifndef USE_ASYNC_LDS
#define USE_ASYNC_LDS 0
#endif

__global__ __launch_bounds__(256) void absmean_partial(
    const float4* __restrict__ W, float* __restrict__ part, size_t n4) {
  float s = 0.f;
  for (size_t i = (size_t)blockIdx.x * 256 + threadIdx.x; i < n4;
       i += (size_t)gridDim.x * 256) {
    float4 w = W[i];
    s += fabsf(w.x) + fabsf(w.y) + fabsf(w.z) + fabsf(w.w);
  }
  __shared__ float red[256];
  red[threadIdx.x] = s;
  __syncthreads();
  for (int st = 128; st > 0; st >>= 1) {
    if ((int)threadIdx.x < st) red[threadIdx.x] += red[threadIdx.x + st];
    __syncthreads();
  }
  if (threadIdx.x == 0) part[blockIdx.x] = red[0];
}

__global__ __launch_bounds__(256) void absmean_final(
    const float* __restrict__ part, int n, float* __restrict__ out,
    float inv_numel) {
  float s = 0.f;
  for (int i = threadIdx.x; i < n; i += 256) s += part[i];
  __shared__ float red[256];
  red[threadIdx.x] = s;
  __syncthreads();
  for (int st = 128; st > 0; st >>= 1) {
    if ((int)threadIdx.x < st) red[threadIdx.x] += red[threadIdx.x + st];
    __syncthreads();
  }
  if (threadIdx.x == 0) out[0] = fmaxf(red[0] * inv_numel, 1e-5f);
}

__global__ __launch_bounds__(256) void quant_w(
    const float4* __restrict__ W, int* __restrict__ Wq,
    const float* __restrict__ scale_p, size_t n4) {
  float inv = 1.0f / scale_p[0];
  for (size_t i = (size_t)blockIdx.x * 256 + threadIdx.x; i < n4;
       i += (size_t)gridDim.x * 256) {
    float4 w = W[i];
    int q0 = (int)fminf(fmaxf(rintf(w.x * inv), -1.f), 1.f);
    int q1 = (int)fminf(fmaxf(rintf(w.y * inv), -1.f), 1.f);
    int q2 = (int)fminf(fmaxf(rintf(w.z * inv), -1.f), 1.f);
    int q3 = (int)fminf(fmaxf(rintf(w.w * inv), -1.f), 1.f);
    Wq[i] = (q0 & 255) | ((q1 & 255) << 8) | ((q2 & 255) << 16) |
            ((q3 & 255) << 24);
  }
}

__global__ __launch_bounds__(256) void quant_rows(
    const float* __restrict__ X, signed char* __restrict__ Xq,
    float* __restrict__ inv, int K) {
  int row = blockIdx.x;
  const float4* xr = (const float4*)(X + (size_t)row * K);
  int K4 = K >> 2;
  float m = 0.f;
  for (int k = threadIdx.x; k < K4; k += 256) {
    float4 v = xr[k];
    m = fmaxf(m, fmaxf(fmaxf(fabsf(v.x), fabsf(v.y)),
                       fmaxf(fabsf(v.z), fabsf(v.w))));
  }
  __shared__ float red[256];
  red[threadIdx.x] = m;
  __syncthreads();
  for (int st = 128; st > 0; st >>= 1) {
    if ((int)threadIdx.x < st)
      red[threadIdx.x] = fmaxf(red[threadIdx.x], red[threadIdx.x + st]);
    __syncthreads();
  }
  float mx = fmaxf(red[0], 1e-5f);
  float scale = 127.f / mx;
  if (threadIdx.x == 0) inv[row] = mx * (1.f / 127.f);
  int* q = (int*)(Xq + (size_t)row * K);
  for (int k = threadIdx.x; k < K4; k += 256) {
    float4 v = xr[k];
    int b0 = (int)fminf(fmaxf(rintf(v.x * scale), -128.f), 127.f);
    int b1 = (int)fminf(fmaxf(rintf(v.y * scale), -128.f), 127.f);
    int b2 = (int)fminf(fmaxf(rintf(v.z * scale), -128.f), 127.f);
    int b3 = (int)fminf(fmaxf(rintf(v.w * scale), -128.f), 127.f);
    q[k] = (b0 & 255) | ((b1 & 255) << 8) | ((b2 & 255) << 16) |
           ((b3 & 255) << 24);
  }
}

__device__ __forceinline__ v8i lds_load_a(const signed char* sA, int r0,
                                          int lane) {
  int row = r0 + (lane & 15);
  int kb = (lane >> 4) << 4;
  const char* base = (const char*)sA + row * LDS_STRIDE + kb;
  int2 p0 = *(const int2*)(base + 0);
  int2 p1 = *(const int2*)(base + 8);
  int2 p2 = *(const int2*)(base + 32);
  int2 p3 = *(const int2*)(base + 40);
  v8i a;
  a[0] = p0.x; a[1] = p0.y; a[2] = p1.x; a[3] = p1.y;
  a[4] = p2.x; a[5] = p2.y; a[6] = p3.x; a[7] = p3.y;
  return a;
}

__device__ __forceinline__ v8i lds_load_b(const signed char* sB, int c0,
                                          int lane) {
  int col = c0 + (lane & 15);
  int koff = (lane >> 4) << 4;
  const char* base = (const char*)sB + col * LDS_STRIDE + koff;
  int4 q0 = *(const int4*)(base + 0);
  int4 q1 = *(const int4*)(base + 32);
  v8i b;
  b[0] = q0.x; b[1] = q0.y; b[2] = q0.z; b[3] = q0.w;
  b[4] = q1.x; b[5] = q1.y; b[6] = q1.z; b[7] = q1.w;
  return b;
}

#if USE_ASYNC_LDS
typedef __attribute__((address_space(1))) v4i* glb_v4i_p;
typedef __attribute__((address_space(3))) v4i* lds_v4i_p;
__device__ __forceinline__ void async_cp16(const signed char* g,
                                           signed char* l) {
  __builtin_amdgcn_global_load_async_to_lds_b128((glb_v4i_p)g, (lds_v4i_p)l,
                                                 0, 0);
}
#endif

__global__ __launch_bounds__(256) void bit_gemm_i8(
    const signed char* __restrict__ A, const signed char* __restrict__ B,
    float* __restrict__ C, const float* __restrict__ x_inv,
    const float* __restrict__ w_scale, int N, int M, int K, int do_relu) {
  __shared__ __attribute__((aligned(16))) signed char sA[2][BT * LDS_STRIDE];
  __shared__ __attribute__((aligned(16))) signed char sB[2][BT * LDS_STRIDE];

  int t = threadIdx.x;
  int wave = t >> 5;
  int lane = t & 31;
  int wr = wave >> 1;
  int wc = wave & 1;
  int n0 = blockIdx.y * BT;
  int m0 = blockIdx.x * BT;

  const signed char* Ap = A + (size_t)n0 * K;
  const signed char* Bp = B + (size_t)m0 * K;

  v8i acc[2][4];
#pragma unroll
  for (int i = 0; i < 2; ++i)
#pragma unroll
    for (int j = 0; j < 4; ++j)
#pragma unroll
      for (int v = 0; v < 8; ++v) acc[i][j][v] = 0;

  int rowLoad = t >> 1;
  int off = (t & 1) * 32;
  int buf = 0;

#if USE_ASYNC_LDS
  {
    const signed char* gA = Ap + (size_t)rowLoad * K + off;
    const signed char* gB = Bp + (size_t)rowLoad * K + off;
    signed char* lA = &sA[0][rowLoad * LDS_STRIDE + off];
    signed char* lB = &sB[0][rowLoad * LDS_STRIDE + off];
    async_cp16(gA, lA);
    async_cp16(gA + 16, lA + 16);
    async_cp16(gB, lB);
    async_cp16(gB + 16, lB + 16);
  }
  for (int k0 = 0; k0 < K; k0 += KT) {
    __builtin_amdgcn_s_wait_asynccnt(0);
    __syncthreads();
    if (k0 + KT < K) {
      const signed char* gA = Ap + (size_t)rowLoad * K + k0 + KT + off;
      const signed char* gB = Bp + (size_t)rowLoad * K + k0 + KT + off;
      signed char* lA = &sA[buf ^ 1][rowLoad * LDS_STRIDE + off];
      signed char* lB = &sB[buf ^ 1][rowLoad * LDS_STRIDE + off];
      async_cp16(gA, lA);
      async_cp16(gA + 16, lA + 16);
      async_cp16(gB, lB);
      async_cp16(gB + 16, lB + 16);
    }
    v8i af[2], bf[4];
#pragma unroll
    for (int i = 0; i < 2; ++i)
      af[i] = lds_load_a(sA[buf], wr * 32 + i * 16, lane);
#pragma unroll
    for (int j = 0; j < 4; ++j)
      bf[j] = lds_load_b(sB[buf], wc * 64 + j * 16, lane);
#pragma unroll
    for (int i = 0; i < 2; ++i)
#pragma unroll
      for (int j = 0; j < 4; ++j)
        acc[i][j] = __builtin_amdgcn_wmma_i32_16x16x64_iu8(
            true, af[i], true, bf[j], acc[i][j], false, false);
    buf ^= 1;
  }
#else
  int4 ra0, ra1, rb0, rb1;
  {
    const int4* ga = (const int4*)(Ap + (size_t)rowLoad * K + off);
    const int4* gb = (const int4*)(Bp + (size_t)rowLoad * K + off);
    ra0 = ga[0]; ra1 = ga[1];
    rb0 = gb[0]; rb1 = gb[1];
  }
  for (int k0 = 0; k0 < K; k0 += KT) {
    *(int4*)(&sA[buf][rowLoad * LDS_STRIDE + off]) = ra0;
    *(int4*)(&sA[buf][rowLoad * LDS_STRIDE + off + 16]) = ra1;
    *(int4*)(&sB[buf][rowLoad * LDS_STRIDE + off]) = rb0;
    *(int4*)(&sB[buf][rowLoad * LDS_STRIDE + off + 16]) = rb1;
    __syncthreads();
    if (k0 + KT < K) {
      const int4* ga = (const int4*)(Ap + (size_t)rowLoad * K + k0 + KT + off);
      const int4* gb = (const int4*)(Bp + (size_t)rowLoad * K + k0 + KT + off);
      ra0 = ga[0]; ra1 = ga[1];
      rb0 = gb[0]; rb1 = gb[1];
      if (k0 + 2 * KT < K) {
        __builtin_prefetch(Ap + (size_t)rowLoad * K + k0 + 2 * KT + off, 0, 1);
        __builtin_prefetch(Bp + (size_t)rowLoad * K + k0 + 2 * KT + off, 0, 1);
      }
    }
    v8i af[2], bf[4];
#pragma unroll
    for (int i = 0; i < 2; ++i)
      af[i] = lds_load_a(sA[buf], wr * 32 + i * 16, lane);
#pragma unroll
    for (int j = 0; j < 4; ++j)
      bf[j] = lds_load_b(sB[buf], wc * 64 + j * 16, lane);
#pragma unroll
    for (int i = 0; i < 2; ++i)
#pragma unroll
      for (int j = 0; j < 4; ++j)
        acc[i][j] = __builtin_amdgcn_wmma_i32_16x16x64_iu8(
            true, af[i], true, bf[j], acc[i][j], false, false);
    buf ^= 1;
  }
#endif

  float wsc = w_scale[0];
#pragma unroll
  for (int i = 0; i < 2; ++i) {
    int rbase = n0 + wr * 32 + i * 16 + ((lane >> 4) << 3);
    float rs[8];
#pragma unroll
    for (int v = 0; v < 8; ++v) rs[v] = wsc * x_inv[rbase + v];
#pragma unroll
    for (int j = 0; j < 4; ++j) {
      int feat = m0 + wc * 64 + j * 16 + (lane & 15);
#pragma unroll
      for (int v = 0; v < 8; ++v) {
        float val = (float)acc[i][j][v] * rs[v];
        if (do_relu) val = fmaxf(val, 0.f);
        C[(size_t)(rbase + v) * M + feat] = val;
      }
    }
  }
}

extern "C" void kernel_launch(void* const* d_in, const int* in_sizes, int n_in,
                              void* d_out, int out_size, void* d_ws,
                              size_t ws_size, hipStream_t stream) {
  const float* x = (const float*)d_in[0];
  const float* W1 = (const float*)d_in[1];
  const float* W2 = (const float*)d_in[2];
  float* out = (float*)d_out;

  const int Ntok = 8192, K1 = 4096, Hd = 16384;
  const size_t nW = (size_t)Hd * K1;

  char* ws = (char*)d_ws;
  size_t off = 0;
  auto carve = [&](size_t bytes) -> void* {
    void* p = ws + off;
    off += (bytes + 255) & ~(size_t)255;
    return p;
  };
  signed char* W1q = (signed char*)carve(nW);
  signed char* W2q = (signed char*)carve(nW);
  signed char* Xq = (signed char*)carve((size_t)Ntok * K1);
  signed char* Hq = (signed char*)carve((size_t)Ntok * Hd);
  float* Hf = (float*)carve((size_t)Ntok * Hd * sizeof(float));
  float* xinv = (float*)carve(Ntok * sizeof(float));
  float* hinv = (float*)carve(Ntok * sizeof(float));
  float* parts = (float*)carve(1024 * sizeof(float));
  float* sW1 = (float*)carve(sizeof(float));
  float* sW2 = (float*)carve(sizeof(float));

  const float inv_numel = 1.0f / (float)nW;

  absmean_partial<<<1024, 256, 0, stream>>>((const float4*)W1, parts, nW / 4);
  absmean_final<<<1, 256, 0, stream>>>(parts, 1024, sW1, inv_numel);
  absmean_partial<<<1024, 256, 0, stream>>>((const float4*)W2, parts, nW / 4);
  absmean_final<<<1, 256, 0, stream>>>(parts, 1024, sW2, inv_numel);

  quant_w<<<2048, 256, 0, stream>>>((const float4*)W1, (int*)W1q, sW1, nW / 4);
  quant_w<<<2048, 256, 0, stream>>>((const float4*)W2, (int*)W2q, sW2, nW / 4);

  quant_rows<<<Ntok, 256, 0, stream>>>(x, Xq, xinv, K1);
  dim3 g1(Hd / BT, Ntok / BT);
  bit_gemm_i8<<<g1, 256, 0, stream>>>(Xq, W1q, Hf, xinv, sW1, Ntok, Hd, K1,
                                       1);

  quant_rows<<<Ntok, 256, 0, stream>>>(Hf, Hq, hinv, Hd);
  dim3 g2(K1 / BT, Ntok / BT);
  bit_gemm_i8<<<g2, 256, 0, stream>>>(Hq, W2q, out, hinv, sW2, Ntok, K1, Hd,
                                       0);
}
